// ONLSTMCell_36687610642958
// MI455X (gfx1250) — matrix operand order repaired and checked by reading, not compiled, not run on hardware
//
#include <hip/hip_runtime.h>
#include <math.h>

// ---------------------------------------------------------------------------
// ON-LSTM cell, MI455X (gfx1250, wave32)
//   B=8192, IN=H=1024, CHUNK=64, NCH=16, GATE=4128, K=2048 (x|hx concat)
// Pipeline:
//   1) pack_a / pack_w : f32 -> bf16 concat into workspace
//   2) gemm_wmma       : gates = Acat @ Wcat^T  (bf16 WMMA, f32 accum,
//                        double-buffered LDS, async global->LDS staging)
//   3) epilogue        : cumsoftmax + gates -> hy, cy, dcf, dcin
// ---------------------------------------------------------------------------

typedef __attribute__((ext_vector_type(8)))  __bf16 v8bf;
typedef __attribute__((ext_vector_type(16))) __bf16 v16bf;
typedef __attribute__((ext_vector_type(8)))  float  v8f;
typedef int vi4 __attribute__((vector_size(16)));   // matches builtin param type

#define B_SZ   8192
#define KDIM   2048
#define GATE   4128
#define HDIM   1024
#define NCH    16

#define TILE_M 128
#define TILE_N 128
#define TILE_K 32
#define LDSP   40   // padded LDS row stride in halfs (80B = 20 dwords -> conflict-free)

#define AS1 __attribute__((address_space(1)))
#define AS3 __attribute__((address_space(3)))

#if __has_builtin(__builtin_amdgcn_global_load_async_to_lds_b128)
  #define USE_ASYNC_LDS 1
#else
  #define USE_ASYNC_LDS 0
#endif

static __device__ __forceinline__ void stage16(const unsigned short* g,
                                               unsigned short* l) {
#if USE_ASYNC_LDS
    __builtin_amdgcn_global_load_async_to_lds_b128(
        (AS1 vi4*)(unsigned long long)g,
        (AS3 vi4*)(unsigned long long)l,
        0, 0);
#else
    *reinterpret_cast<uint4*>(l) = *reinterpret_cast<const uint4*>(g);
#endif
}

static __device__ __forceinline__ void wait_stage() {
#if USE_ASYNC_LDS
  #if __has_builtin(__builtin_amdgcn_s_wait_asynccnt)
    __builtin_amdgcn_s_wait_asynccnt(0);
  #else
    asm volatile("s_wait_asynccnt 0x0" ::: "memory");
  #endif
#endif
}

static __device__ __forceinline__ unsigned short f2bf(float f) {
    unsigned int u = __float_as_uint(f);
    unsigned int r = (u + 0x7FFFu + ((u >> 16) & 1u)) >> 16;   // RNE
    return (unsigned short)r;
}

static __device__ __forceinline__ float sigmoidf(float x) {
    return 1.0f / (1.0f + expf(-x));
}

// ---------------------------------------------------------------------------
// 1) Pack kernels
// ---------------------------------------------------------------------------
__global__ __launch_bounds__(256)
void pack_a(const float* __restrict__ x, const float* __restrict__ hx,
            unsigned short* __restrict__ Acat) {
    int idx = blockIdx.x * 256 + threadIdx.x;
    int b = idx >> 11;
    int k = idx & 2047;
    float v = (k < HDIM) ? x[b * HDIM + k] : hx[b * HDIM + (k - HDIM)];
    Acat[idx] = f2bf(v);
}

__global__ __launch_bounds__(256)
void pack_w(const float* __restrict__ wih, const float* __restrict__ whh,
            unsigned short* __restrict__ Wcat) {
    int idx = blockIdx.x * 256 + threadIdx.x;
    int g = idx >> 11;
    int k = idx & 2047;
    float v = (k < HDIM) ? wih[g * HDIM + k] : whh[g * HDIM + (k - HDIM)];
    Wcat[idx] = f2bf(v);
}

// ---------------------------------------------------------------------------
// 2) bf16 WMMA GEMM:  gates[8192, 4128] = Acat[M,K] * Wcat[N,K]^T
//    256 threads = 8 waves, tile 128M x 128N, K double-buffered 32-wide.
//    Wave (wm,wn) in 2x4 grid owns 64M x 32N = 4x2 accumulators of 16x16.
// ---------------------------------------------------------------------------
__global__ __launch_bounds__(256)
void gemm_wmma(const unsigned short* __restrict__ Acat,
               const unsigned short* __restrict__ Wcat,
               float* __restrict__ gates)
{
    __shared__ alignas(16) unsigned short As[2][TILE_M * LDSP];
    __shared__ alignas(16) unsigned short Bs[2][TILE_N * LDSP];

    const int tid  = threadIdx.x;
    const int lane = tid & 31;
    const int wave = tid >> 5;
    const int wm   = wave & 1;     // 0..1  -> 64-row slab
    const int wn   = wave >> 1;    // 0..3  -> 32-col slab
    const int m0   = blockIdx.y * TILE_M;
    const int n0   = blockIdx.x * TILE_N;

    const int mrow = lane & 15;    // M (A frag) / N (B frag) index within tile
    const int half = lane >> 4;    // K-half selector

    v8f acc[4][2];
    {
        v8f z = {};
        #pragma unroll
        for (int mt = 0; mt < 4; ++mt)
            #pragma unroll
            for (int nt = 0; nt < 2; ++nt)
                acc[mt][nt] = z;
    }

    // stage one 128x32 A tile + 128x32 W tile into LDS buffer `buf`
    auto stage_tiles = [&](int k0, int buf) {
        #pragma unroll
        for (int i = 0; i < 2; ++i) {
            int cid = tid + i * 256;           // 0..511
            int r   = cid >> 2;                // row 0..127
            int cc  = (cid & 3) << 3;          // col 0,8,16,24 (halfs)
            stage16(&Acat[(m0 + r) * KDIM + k0 + cc], &As[buf][r * LDSP + cc]);
        }
        #pragma unroll
        for (int i = 0; i < 2; ++i) {
            int cid = tid + i * 256;
            int r   = cid >> 2;
            int cc  = (cid & 3) << 3;
            int g   = n0 + r; if (g > GATE - 1) g = GATE - 1;
            stage16(&Wcat[g * KDIM + k0 + cc], &Bs[buf][r * LDSP + cc]);
        }
    };

    stage_tiles(0, 0);

    const int NK = KDIM / TILE_K;              // 64
    for (int kt = 0; kt < NK; ++kt) {
        const int buf = kt & 1;
        wait_stage();          // this wave's async loads into `buf` complete
        __syncthreads();       // everyone's loads visible; prev compute done

        if (kt + 1 < NK) {
            stage_tiles((kt + 1) * TILE_K, buf ^ 1);
        }
#if !USE_ASYNC_LDS
        // sync fallback: prefetch keeps some overlap
        if (kt + 1 < NK) {
            int r  = tid >> 1;
            int co = (tid & 1) * 16;
            __builtin_prefetch(&Acat[(m0 + r) * KDIM + (kt + 1) * TILE_K + co], 0, 0);
        }
#endif

        v16bf afrag[4];
        v16bf bfrag[2];
        #pragma unroll
        for (int mt = 0; mt < 4; ++mt) {
            int row = wm * 64 + mt * 16 + mrow;
            const v8bf* lo = reinterpret_cast<const v8bf*>(
                &As[buf][row * LDSP + half * 8]);
            const v8bf* hi = reinterpret_cast<const v8bf*>(
                &As[buf][row * LDSP + half * 8 + 16]);
            afrag[mt] = __builtin_shufflevector(*lo, *hi,
                0,1,2,3,4,5,6,7,8,9,10,11,12,13,14,15);
        }
        #pragma unroll
        for (int nt = 0; nt < 2; ++nt) {
            int row = wn * 32 + nt * 16 + mrow;
            const v8bf* lo = reinterpret_cast<const v8bf*>(
                &Bs[buf][row * LDSP + half * 8]);
            const v8bf* hi = reinterpret_cast<const v8bf*>(
                &Bs[buf][row * LDSP + half * 8 + 16]);
            bfrag[nt] = __builtin_shufflevector(*lo, *hi,
                0,1,2,3,4,5,6,7,8,9,10,11,12,13,14,15);
        }
        #pragma unroll
        for (int mt = 0; mt < 4; ++mt)
            #pragma unroll
            for (int nt = 0; nt < 2; ++nt)
                acc[mt][nt] = __builtin_amdgcn_wmma_f32_16x16x32_bf16(
                    false, afrag[mt], false, bfrag[nt],
                    (short)0, acc[mt][nt], false, false);
    }

    // store: C/D layout: vgpr r -> M = r + (lane/16)*8, N = lane%16
    const int col  = lane & 15;
    const int rofs = (lane >> 4) * 8;
    #pragma unroll
    for (int mt = 0; mt < 4; ++mt) {
        #pragma unroll
        for (int nt = 0; nt < 2; ++nt) {
            int n = n0 + wn * 32 + nt * 16 + col;
            if (n < GATE) {
                int mbase = m0 + wm * 64 + mt * 16 + rofs;
                #pragma unroll
                for (int r = 0; r < 8; ++r)
                    gates[(mbase + r) * GATE + n] = acc[mt][nt][r];
            }
        }
    }
}

// ---------------------------------------------------------------------------
// 3) Epilogue: one block (256 thr) per sample.
// ---------------------------------------------------------------------------
__global__ __launch_bounds__(256)
void onlstm_epilogue(const float* __restrict__ gates,
                     const float* __restrict__ b_ih,
                     const float* __restrict__ b_hh,
                     const float* __restrict__ cx,
                     float* __restrict__ hy, float* __restrict__ cy,
                     float* __restrict__ dcf, float* __restrict__ dcin)
{
    __shared__ float ci_s[NCH];
    __shared__ float cf_s[NCH];
    __shared__ float raw[2 * NCH];

    const int b = blockIdx.x;
    const int t = threadIdx.x;
    const float* grow = gates + b * GATE;

    if (t < 2 * NCH) raw[t] = grow[t] + b_ih[t] + b_hh[t];
    __syncthreads();

    if (t < 2) {
        // t==0: cingate = 1 - cumsoftmax(raw[0:16]); t==1: cforget = cumsoftmax(raw[16:32])
        const float* r = raw + t * NCH;
        float mx = r[0];
        #pragma unroll
        for (int i = 1; i < NCH; ++i) mx = fmaxf(mx, r[i]);
        float e[NCH], s = 0.0f;
        #pragma unroll
        for (int i = 0; i < NCH; ++i) { e[i] = expf(r[i] - mx); s += e[i]; }
        float inv = 1.0f / s, cs = 0.0f, acc = 0.0f;
        float* dst = (t == 0) ? ci_s : cf_s;
        #pragma unroll
        for (int i = 0; i < NCH; ++i) {
            cs += e[i] * inv;
            float v = (t == 0) ? (1.0f - cs) : cs;
            dst[i] = v;
            acc += v;
        }
        if (t == 0) dcin[b] = acc * (1.0f / NCH);
        else        dcf[b]  = 1.0f - acc * (1.0f / NCH);
    }
    __syncthreads();

    for (int i = t; i < HDIM; i += 256) {
        int j = i >> 6;  // chunk index
        float og = sigmoidf(grow[32 +        i] + b_ih[32 +        i] + b_hh[32 +        i]);
        float cl = tanhf  (grow[32 + 1024 + i] + b_ih[32 + 1024 + i] + b_hh[32 + 1024 + i]);
        float ig = sigmoidf(grow[32 + 2048 + i] + b_ih[32 + 2048 + i] + b_hh[32 + 2048 + i]);
        float fg = sigmoidf(grow[32 + 3072 + i] + b_ih[32 + 3072 + i] + b_hh[32 + 3072 + i]);
        float ci = ci_s[j], cf = cf_s[j];
        float ov = cf * ci;
        float fgate = fg * ov + (cf - ov);
        float igate = ig * ov + (ci - ov);
        float c = fgate * cx[b * HDIM + i] + igate * cl;
        cy[b * HDIM + i] = c;
        hy[b * HDIM + i] = og * tanhf(c);
    }
}

// ---------------------------------------------------------------------------
extern "C" void kernel_launch(void* const* d_in, const int* in_sizes, int n_in,
                              void* d_out, int out_size, void* d_ws, size_t ws_size,
                              hipStream_t stream) {
    const float* x    = (const float*)d_in[0];
    const float* hx   = (const float*)d_in[1];
    const float* cx   = (const float*)d_in[2];
    const float* W_ih = (const float*)d_in[3];
    const float* b_ih = (const float*)d_in[4];
    const float* W_hh = (const float*)d_in[5];
    const float* b_hh = (const float*)d_in[6];

    float* out  = (float*)d_out;
    float* hy   = out;
    float* cy   = out + (size_t)B_SZ * HDIM;
    float* dcf  = out + (size_t)2 * B_SZ * HDIM;
    float* dcin = dcf + B_SZ;

    char* ws = (char*)d_ws;
    unsigned short* Acat = (unsigned short*)ws;                       // 32 MiB
    unsigned short* Wcat = (unsigned short*)(ws + 33554432);          // ~16.1 MiB
    float*          gate = (float*)(ws + 50462720);                   // ~129 MiB

    pack_a<<<(B_SZ * KDIM) / 256, 256, 0, stream>>>(x, hx, Acat);
    pack_w<<<(GATE * KDIM) / 256, 256, 0, stream>>>(W_ih, W_hh, Wcat);

    dim3 grid((GATE + TILE_N - 1) / TILE_N, B_SZ / TILE_M);           // 33 x 64
    gemm_wmma<<<grid, 256, 0, stream>>>(Acat, Wcat, gate);

    onlstm_epilogue<<<B_SZ, 256, 0, stream>>>(gate, b_ih, b_hh, cx,
                                              hy, cy, dcf, dcin);
}
